// BaselineClassifier_2877628088443
// MI455X (gfx1250) — matrix operand order repaired and checked by reading, not compiled, not run on hardware
//
#include <hip/hip_runtime.h>
#include <hip/hip_bf16.h>

typedef __attribute__((ext_vector_type(16))) _Float16 v16h;
typedef __attribute__((ext_vector_type(8)))  _Float16 v8h;
typedef __attribute__((ext_vector_type(8)))  float    v8f;

#define HID 64
#define IN_DIM 34
#define EA_STRIDE 48   // halves per row (34 real + pad); 96B rows, 16B aligned
#define NUM_GRAPHS 64
#define NUM_CLASSES 10

// ---------------- WMMA helpers (CDNA5 gfx1250, wave32) ----------------

__device__ __forceinline__ v8f wmma_f16(v16h a, v16h b, v8f c) {
  // (neg_a, A, neg_b, B, c_mod, C, reuse_a, reuse_b)
  return __builtin_amdgcn_wmma_f32_16x16x32_f16(false, a, false, b, (short)0, c, false, false);
}

// A-matrix 16x32 f16 fragment from row-major f16 memory (16 rows, strideH halves).
// ISA layout: lane&15 = M row; lanes<16 halves0-7 = K0-7, halves8-15 = K16-23;
// lanes>=16 halves0-7 = K8-15, halves8-15 = K24-31.  -> two 16B chunks per lane.
__device__ __forceinline__ v16h load_afrag(const _Float16* p, int strideH, int lane, int colBase) {
  const int m  = lane & 15;
  const int hi = (lane >> 4) & 1;
  const _Float16* rp = p + (size_t)m * strideH + colBase + hi * 8;
  v8h a = *(const v8h*)(rp);
  v8h b = *(const v8h*)(rp + 16);
  v16h f;
#pragma unroll
  for (int i = 0; i < 8; ++i) { f[i] = a[i]; f[8 + i] = b[i]; }
  return f;
}

// B-matrix 32x16 f16 fragment from LDS 64x64 row-major f16 weights.
// Layout: lane&15 = N col; lanes<16 halves0-7 = K0-7, halves8-15 = K16-23; lanes>=16 halves0-7 = K8-15, halves8-15 = K24-31.
__device__ __forceinline__ v16h load_bfrag(const _Float16* W, int lane, int kBase, int nBase) {
  const int n = lane & 15;
  const int kOff = ((lane >> 4) & 1) * 8;
  v16h f;
#pragma unroll
  for (int h = 0; h < 16; ++h) f[h] = W[(size_t)(kBase + kOff + (h & 7) + ((h >> 3) << 4)) * 64 + nBase + n];
  return f;
}

// ---------------- K1: edge feature assembly + self-loop accumulation ----------------

__global__ __launch_bounds__(256) void k1_edge_assemble(
    const int* __restrict__ edge_index, const int* __restrict__ dst_ports,
    const int* __restrict__ tcp_flags, const float* __restrict__ edge_attr,
    const float* __restrict__ emb_port, const float* __restrict__ emb_flags,
    _Float16* __restrict__ ea, int* __restrict__ dstfull,
    float* __restrict__ loopsum, int* __restrict__ cnt, int E)
{
  int e = blockIdx.x * blockDim.x + threadIdx.x;
  if (e >= E) return;
  int c = edge_index[E + e];
  dstfull[e] = c;
  atomicAdd(&cnt[c], 1);
  float f[IN_DIM];
  const float* ap = edge_attr + (size_t)e * 16;
#pragma unroll
  for (int j = 0; j < 16; ++j) f[j] = ap[j];
  const float* pp = emb_port + (size_t)dst_ports[e] * 16;
#pragma unroll
  for (int j = 0; j < 16; ++j) f[16 + j] = pp[j];
  const float* fp = emb_flags + (size_t)tcp_flags[e] * 2;
  f[32] = fp[0]; f[33] = fp[1];
  _Float16* row = ea + (size_t)e * EA_STRIDE;
#pragma unroll
  for (int j = 0; j < IN_DIM; ++j) row[j] = (_Float16)f[j];
#pragma unroll
  for (int j = IN_DIM; j < EA_STRIDE; ++j) row[j] = (_Float16)0.f;
  float* ls = loopsum + (size_t)c * IN_DIM;
#pragma unroll
  for (int j = 0; j < IN_DIM; ++j) atomicAdd(&ls[j], f[j]);
}

// ---------------- K2: self-loop rows, deg ----------------

__global__ __launch_bounds__(256) void k2_selfloop(
    const int* __restrict__ cnt, const float* __restrict__ loopsum,
    _Float16* __restrict__ ea, int* __restrict__ dstfull,
    float* __restrict__ deg, int E, int N)
{
  int n = blockIdx.x * blockDim.x + threadIdx.x;
  if (n >= N) return;
  int c = cnt[n];
  float inv = 1.f / fmaxf((float)c, 1.f);
  _Float16* row = ea + (size_t)(E + n) * EA_STRIDE;
  const float* ls = loopsum + (size_t)n * IN_DIM;
#pragma unroll
  for (int j = 0; j < IN_DIM; ++j) row[j] = (_Float16)(ls[j] * inv);
#pragma unroll
  for (int j = IN_DIM; j < EA_STRIDE; ++j) row[j] = (_Float16)0.f;
  dstfull[E + n] = n;
  deg[n] = (float)c + 1.f;  // +1 self loop
}

// ---------------- K3: fused edge-MLP (WMMA) + scatter into agg ----------------

__global__ __launch_bounds__(128) void k3_mlp_scatter(
    const _Float16* __restrict__ ea, const int* __restrict__ dstfull,
    const float* __restrict__ W1, const float* __restrict__ b1,
    const float* __restrict__ W2, const float* __restrict__ b2,
    float* __restrict__ agg, int totalRows, int nTile64)
{
  __shared__ __align__(16) _Float16 sW1[64 * 64];
  __shared__ __align__(16) _Float16 sW2[64 * 64];
  __shared__ float sb1[64];
  __shared__ float sb2[64];
  __shared__ __align__(16) _Float16 sH[4][16 * 64];  // per-wave hidden tile

  const int tid = threadIdx.x;
  for (int i = tid; i < 64 * 64; i += 128) {
    int r = i >> 6, c = i & 63;
    sW1[i] = (r < IN_DIM) ? (_Float16)W1[r * 64 + c] : (_Float16)0.f;
    sW2[i] = (_Float16)W2[i];
  }
  if (tid < 64) { sb1[tid] = b1[tid]; sb2[tid] = b2[tid]; }
  __syncthreads();

  const int lane = tid & 31;
  const int wave = tid >> 5;
  const int nIdx = lane & 15;
  const int hi   = (lane >> 4) & 1;

  // weight fragments live in registers for the whole kernel
  v16h fW1[2][4], fW2[2][4];
#pragma unroll
  for (int k = 0; k < 2; ++k)
#pragma unroll
    for (int n = 0; n < 4; ++n) {
      fW1[k][n] = load_bfrag(sW1, lane, k * 32, n * 16);
      fW2[k][n] = load_bfrag(sW2, lane, k * 32, n * 16);
    }

  _Float16* hbuf = sH[wave];

  for (int t = blockIdx.x; t < nTile64; t += gridDim.x) {
    const int base = t * 64 + wave * 16;
    const _Float16* ap = ea + (size_t)base * EA_STRIDE;
    // layer 1: A = 16x64 features (k-step1 chunkB reads padded/junk cols * zero W1 rows)
    v16h a0 = load_afrag(ap, EA_STRIDE, lane, 0);
    v16h a1 = load_afrag(ap, EA_STRIDE, lane, 32);
#pragma unroll
    for (int n = 0; n < 4; ++n) {
      float bv = sb1[n * 16 + nIdx];
      v8f acc;
#pragma unroll
      for (int r = 0; r < 8; ++r) acc[r] = bv;
      acc = wmma_f16(a0, fW1[0][n], acc);
      acc = wmma_f16(a1, fW1[1][n], acc);
#pragma unroll
      for (int r = 0; r < 8; ++r) {
        float v = acc[r] > 0.f ? acc[r] : 0.f;  // relu
        hbuf[(size_t)(r + hi * 8) * 64 + n * 16 + nIdx] = (_Float16)v;
      }
    }
    // re-enter A-fragment layout for layer 2 (same-wave LDS RAW: dscnt handled by compiler)
    v16h h0 = load_afrag(hbuf, 64, lane, 0);
    v16h h1 = load_afrag(hbuf, 64, lane, 32);

    int dcache[8];
#pragma unroll
    for (int r = 0; r < 8; ++r) {
      int grow = base + r + hi * 8;
      dcache[r] = (grow < totalRows) ? dstfull[grow] : -1;
    }
#pragma unroll
    for (int n = 0; n < 4; ++n) {
      float bv = sb2[n * 16 + nIdx];
      v8f acc;
#pragma unroll
      for (int r = 0; r < 8; ++r) acc[r] = bv;
      acc = wmma_f16(h0, fW2[0][n], acc);
      acc = wmma_f16(h1, fW2[1][n], acc);
#pragma unroll
      for (int r = 0; r < 8; ++r) {
        if (dcache[r] >= 0)
          atomicAdd(&agg[(size_t)dcache[r] * HID + n * 16 + nIdx], acc[r]);
      }
    }
  }
}

// ---------------- K4: m-bar = agg/deg ; x1 = m-bar ----------------

__global__ __launch_bounds__(256) void k4_finalize_mbar(
    const float* __restrict__ deg, float* __restrict__ agg,
    float* __restrict__ x, int N)
{
  int i = blockIdx.x * blockDim.x + threadIdx.x;
  if (i >= N * HID) return;
  float v = agg[i] / deg[i >> 6];
  agg[i] = v;   // m-bar
  x[i] = v;     // x after layer 1
}

// ---------------- K5a: scatter x[src] -> xacc[dst] over real edges ----------------

__global__ __launch_bounds__(256) void k5a_prop_scatter(
    const int* __restrict__ edge_index, const float* __restrict__ x,
    float* __restrict__ xacc, int E)
{
  int t = blockIdx.x * blockDim.x + threadIdx.x;
  int e = t >> 4;
  if (e >= E) return;
  int part = t & 15;
  int r = edge_index[e];
  int c = edge_index[E + e];
  float4 v = *(const float4*)(x + (size_t)r * HID + part * 4);
  float* d = xacc + (size_t)c * HID + part * 4;
  atomicAdd(d + 0, v.x);
  atomicAdd(d + 1, v.y);
  atomicAdd(d + 2, v.z);
  atomicAdd(d + 3, v.w);
}

// ---------------- K5b: x_out = (xacc + x_self)/deg + mbar ----------------

__global__ __launch_bounds__(256) void k5b_prop_finalize(
    const float* __restrict__ xacc, const float* __restrict__ xin,
    const float* __restrict__ mbar, const float* __restrict__ deg,
    float* __restrict__ xout, int N)
{
  int i = blockIdx.x * blockDim.x + threadIdx.x;
  if (i >= N * HID) return;
  xout[i] = (xacc[i] + xin[i]) / deg[i >> 6] + mbar[i];
}

// ---------------- K6: per-graph mean/max pooling ----------------

__global__ __launch_bounds__(256) void k6_pool(
    const float* __restrict__ x, const int* __restrict__ batch,
    float* __restrict__ meanacc, unsigned* __restrict__ maxenc,
    float* __restrict__ gcnt, int N)
{
  int i = blockIdx.x * blockDim.x + threadIdx.x;
  if (i >= N * HID) return;
  int n = i >> 6, c = i & 63;
  int g = batch[n];
  float v = x[i];
  atomicAdd(&meanacc[g * HID + c], v);
  unsigned b = __float_as_uint(v);
  unsigned enc = (b & 0x80000000u) ? ~b : (b | 0x80000000u);  // order-preserving
  atomicMax(&maxenc[g * HID + c], enc);
  if (c == 0) atomicAdd(&gcnt[g], 1.f);
}

// ---------------- K7: classifier MLP (tiny, fp32 scalar) ----------------

__global__ __launch_bounds__(64) void k7_classifier(
    const float* __restrict__ meanacc, const unsigned* __restrict__ maxenc,
    const float* __restrict__ gcnt,
    const float* __restrict__ CW1, const float* __restrict__ Cb1,
    const float* __restrict__ CW2, const float* __restrict__ Cb2,
    float* __restrict__ out)
{
  int i = threadIdx.x;  // graph id, single block of 64
  float pooled[2 * HID];
  float inv = 1.f / fmaxf(gcnt[i], 1.f);
#pragma unroll
  for (int c = 0; c < HID; ++c) pooled[c] = meanacc[i * HID + c] * inv;
#pragma unroll
  for (int c = 0; c < HID; ++c) {
    unsigned u = maxenc[i * HID + c];
    unsigned b = (u & 0x80000000u) ? (u & 0x7FFFFFFFu) : ~u;
    pooled[HID + c] = __uint_as_float(b);
  }
  float h[HID];
  for (int j = 0; j < HID; ++j) {
    float s = Cb1[j];
#pragma unroll 8
    for (int k = 0; k < 2 * HID; ++k) s += pooled[k] * CW1[k * HID + j];
    h[j] = fmaxf(s, 0.f);
  }
  for (int c = 0; c < NUM_CLASSES; ++c) {
    float s = Cb2[c];
#pragma unroll 8
    for (int j = 0; j < HID; ++j) s += h[j] * CW2[j * NUM_CLASSES + c];
    out[i * NUM_CLASSES + c] = s;
  }
}

// ---------------- host launcher ----------------

static inline size_t alignUp256(size_t x) { return (x + 255) & ~(size_t)255; }

extern "C" void kernel_launch(void* const* d_in, const int* in_sizes, int n_in,
                              void* d_out, int out_size, void* d_ws, size_t ws_size,
                              hipStream_t stream) {
  const int*   edge_index = (const int*)d_in[0];
  const int*   dst_ports  = (const int*)d_in[1];
  const int*   tcp_flags  = (const int*)d_in[2];
  const float* edge_attr  = (const float*)d_in[3];
  const int*   batch      = (const int*)d_in[4];
  const float* emb_port   = (const float*)d_in[5];
  const float* emb_flags  = (const float*)d_in[6];
  const float* W1  = (const float*)d_in[7];
  const float* b1  = (const float*)d_in[8];
  const float* W2  = (const float*)d_in[9];
  const float* b2  = (const float*)d_in[10];
  const float* CW1 = (const float*)d_in[11];
  const float* Cb1 = (const float*)d_in[12];
  const float* CW2 = (const float*)d_in[13];
  const float* Cb2 = (const float*)d_in[14];
  float* out = (float*)d_out;

  const int E = in_sizes[1];       // dst_ports length
  const int N = in_sizes[4];       // batch length
  const int total = E + N;
  const int nT64 = (total + 63) / 64;
  const size_t rpad = (size_t)nT64 * 64 + 64;

  // workspace carve-up
  char* w = (char*)d_ws;
  size_t off = 0;
  auto alloc = [&](size_t bytes) -> void* {
    void* p = w + off;
    off = alignUp256(off + bytes);
    return p;
  };
  _Float16* ea      = (_Float16*)alloc(rpad * EA_STRIDE * 2);
  int*      dstfull = (int*)     alloc((size_t)total * 4);
  float*    loopsum = (float*)   alloc((size_t)N * IN_DIM * 4);
  int*      cnt     = (int*)     alloc((size_t)N * 4);
  float*    deg     = (float*)   alloc((size_t)N * 4);
  float*    agg     = (float*)   alloc((size_t)N * HID * 4);   // -> becomes m-bar
  float*    xA      = (float*)   alloc((size_t)N * HID * 4);
  float*    xB      = (float*)   alloc((size_t)N * HID * 4);
  float*    xacc    = (float*)   alloc((size_t)N * HID * 4);
  float*    meanacc = (float*)   alloc((size_t)NUM_GRAPHS * HID * 4);
  unsigned* maxenc  = (unsigned*)alloc((size_t)NUM_GRAPHS * HID * 4);
  float*    gcnt    = (float*)   alloc((size_t)NUM_GRAPHS * 4);
  (void)ws_size;

  // zero accumulators (graph-capturable async memsets)
  hipMemsetAsync(loopsum, 0, (size_t)N * IN_DIM * 4, stream);
  hipMemsetAsync(cnt,     0, (size_t)N * 4, stream);
  hipMemsetAsync(agg,     0, (size_t)N * HID * 4, stream);
  hipMemsetAsync(meanacc, 0, (size_t)NUM_GRAPHS * HID * 4, stream);
  hipMemsetAsync(maxenc,  0, (size_t)NUM_GRAPHS * HID * 4, stream);
  hipMemsetAsync(gcnt,    0, (size_t)NUM_GRAPHS * 4, stream);

  k1_edge_assemble<<<(E + 255) / 256, 256, 0, stream>>>(
      edge_index, dst_ports, tcp_flags, edge_attr, emb_port, emb_flags,
      ea, dstfull, loopsum, cnt, E);

  k2_selfloop<<<(N + 255) / 256, 256, 0, stream>>>(
      cnt, loopsum, ea, dstfull, deg, E, N);

  int blocks3 = nT64 < 4096 ? nT64 : 4096;
  k3_mlp_scatter<<<blocks3, 128, 0, stream>>>(
      ea, dstfull, W1, b1, W2, b2, agg, total, nT64);

  k4_finalize_mbar<<<(N * HID + 255) / 256, 256, 0, stream>>>(deg, agg, xA, N);

  // propagation layer 2: xA -> xB
  hipMemsetAsync(xacc, 0, (size_t)N * HID * 4, stream);
  k5a_prop_scatter<<<((size_t)E * 16 + 255) / 256, 256, 0, stream>>>(edge_index, xA, xacc, E);
  k5b_prop_finalize<<<(N * HID + 255) / 256, 256, 0, stream>>>(xacc, xA, agg, deg, xB, N);

  // propagation layer 3: xB -> xA
  hipMemsetAsync(xacc, 0, (size_t)N * HID * 4, stream);
  k5a_prop_scatter<<<((size_t)E * 16 + 255) / 256, 256, 0, stream>>>(edge_index, xB, xacc, E);
  k5b_prop_finalize<<<(N * HID + 255) / 256, 256, 0, stream>>>(xacc, xB, agg, deg, xA, N);

  k6_pool<<<(N * HID + 255) / 256, 256, 0, stream>>>(xA, batch, meanacc, maxenc, gcnt, N);

  k7_classifier<<<1, 64, 0, stream>>>(meanacc, maxenc, gcnt, CW1, Cb1, CW2, Cb2, out);
}
